// InteractionTargetDiversityScorer_47854525612395
// MI455X (gfx1250) — matrix operand order repaired and checked by reading, not compiled, not run on hardware
//
#include <hip/hip_runtime.h>
#include <hip/hip_bf16.h>

typedef __attribute__((ext_vector_type(16))) _Float16 v16h;
typedef __attribute__((ext_vector_type(8)))  float    v8f;

#define MDIM 128
#define RDIM 32
#define DDIM 128
#define NHS  132   // padded LDS stride (halves) for nhT[d][m]

__device__ __forceinline__ float wave_sum(float v) {
#pragma unroll
  for (int m = 16; m > 0; m >>= 1) v += __shfl_xor(v, m, 32);
  return v;
}
__device__ __forceinline__ float wave_max(float v) {
#pragma unroll
  for (int m = 16; m > 0; m >>= 1) v = fmaxf(v, __shfl_xor(v, m, 32));
  return v;
}

__global__ __launch_bounds__(256)
void itds_kernel(const float* __restrict__ ego_adj,
                 const int*   __restrict__ ego_rel,
                 const float* __restrict__ alpha_r,
                 const float* __restrict__ tau_max,
                 const float* __restrict__ p_i_r,
                 const float* __restrict__ h_raw,
                 const unsigned char* __restrict__ adj_mask,
                 const float* __restrict__ w_r,
                 const float* __restrict__ w_c,
                 float* __restrict__ out)
{
  __shared__ _Float16 nhT[DDIM * NHS];   // nhT[d*NHS + m], f16 normalized rows (transposed)
  __shared__ float valf[MDIM];           // 0/1 validity
  __shared__ int   tval[MDIM];           // clamped type if valid, else -1
  __shared__ float alphas[RDIM];
  __shared__ float gsq[RDIM];            // ||sum_{m in r} nh_m||^2
  __shared__ float snorm[RDIM];          // sum_{m in r} ||nh_m||^2 (f16-rounded)
  __shared__ float cntf[RDIM];           // c_r
  __shared__ float red[8];
  __shared__ float s_num;

  const int b    = blockIdx.x;
  const int tid  = threadIdx.x;
  const int wave = tid >> 5;
  const int lane = tid & 31;

  // ---------------- phase 0: small tables ----------------
  if (tid < MDIM) {
    float v = adj_mask[(size_t)b * MDIM + tid] ? 1.0f : 0.0f;
    valf[tid] = v;
    int t = ego_rel[(size_t)b * MDIM * MDIM + tid];       // s_types = ego_rel[b,0,:]
    t = t < 0 ? 0 : (t > RDIM - 1 ? RDIM - 1 : t);
    tval[tid] = (v > 0.5f) ? t : -1;
  } else if (tid < 160) {
    alphas[tid - 128] = alpha_r[tid - 128];
  } else if (tid < 192) {
    gsq[tid - 160] = 0.0f;
  } else if (tid < 224) {
    snorm[tid - 192] = 0.0f;
  } else {
    cntf[tid - 224] = 0.0f;
  }
  __syncthreads();

  // ---------------- phase 1: numerator = sum ego_adj*alpha[rel]*mask_m*mask_n ----------------
  {
    const float4* pa = reinterpret_cast<const float4*>(ego_adj + (size_t)b * MDIM * MDIM);
    const int4*   pr = reinterpret_cast<const int4*>  (ego_rel + (size_t)b * MDIM * MDIM);
    float acc = 0.0f;
#pragma unroll 4
    for (int it = 0; it < 16; ++it) {
      int idx4 = it * 256 + tid;      // 0..4095 float4 chunks
      int base = idx4 << 2;           // element index in [0,16384)
      int m = base >> 7;
      int n = base & 127;
      float4 a = pa[idx4];
      int4   r = pr[idx4];
      int r0 = min(max(r.x, 0), 31), r1 = min(max(r.y, 0), 31);
      int r2 = min(max(r.z, 0), 31), r3 = min(max(r.w, 0), 31);
      float s = a.x * alphas[r0] * valf[n + 0]
              + a.y * alphas[r1] * valf[n + 1]
              + a.z * alphas[r2] * valf[n + 2]
              + a.w * alphas[r3] * valf[n + 3];
      acc += valf[m] * s;
    }
    acc = wave_sum(acc);
    if (lane == 0) red[wave] = acc;
  }
  __syncthreads();
  if (tid == 0) {
    float s = 0.0f;
#pragma unroll
    for (int i = 0; i < 8; ++i) s += red[i];
    s_num = s;
  }

  // ---------------- phase 2: normalize rows -> f16 nhT, per-type stats ----------------
#pragma unroll 1
  for (int it = 0; it < 16; ++it) {
    int m = it * 8 + wave;
    const float4* ph = reinterpret_cast<const float4*>(h_raw + ((size_t)b * MDIM + m) * DDIM);
    float4 h = ph[lane];
    float ss = h.x * h.x + h.y * h.y + h.z * h.z + h.w * h.w;
    ss = wave_sum(ss);
    float inv = rsqrtf(fmaxf(ss, 1e-12f));
    _Float16 q0 = (_Float16)(h.x * inv);
    _Float16 q1 = (_Float16)(h.y * inv);
    _Float16 q2 = (_Float16)(h.z * inv);
    _Float16 q3 = (_Float16)(h.w * inv);
    int d0 = lane * 4;
    nhT[(d0 + 0) * NHS + m] = q0;
    nhT[(d0 + 1) * NHS + m] = q1;
    nhT[(d0 + 2) * NHS + m] = q2;
    nhT[(d0 + 3) * NHS + m] = q3;
    // row norm of the f16-rounded vector (so the diagonal cancels exactly vs gsq)
    float f0 = (float)q0, f1 = (float)q1, f2 = (float)q2, f3 = (float)q3;
    float rn = wave_sum(f0 * f0 + f1 * f1 + f2 * f2 + f3 * f3);
    if (lane == 0) {
      int t = tval[m];
      if (t >= 0) {
        atomicAdd(&snorm[t], rn);
        atomicAdd(&cntf[t], 1.0f);
      }
    }
  }
  __syncthreads();

  // ---------------- phase 3: G = oh^T * nh via WMMA, accumulate ||G_r||^2 ----------------
  {
    const int j  = lane & 15;     // output column within tile / A row index
    const int hi = lane >> 4;
    const int d0 = wave * 16;     // this wave's 16-wide d slice
    v8f acc0 = {};                // r-tile 0 (r = 0..15)
    v8f acc1 = {};                // r-tile 1 (r = 16..31)
#pragma unroll 1
    for (int kc = 0; kc < 4; ++kc) {
      const int mb = kc * 32;     // K (=m) chunk base
      v16h bv, a0, a1;
      // B (32x16 f16): lanes 0-15 col N=lane hold K=0..15; lanes 16-31 hold K=16..31
      const int mB = mb + hi * 16;
#pragma unroll
      for (int e = 0; e < 16; ++e)
        bv[e] = nhT[(d0 + j) * NHS + mB + e];
      // A (16x32 f16): lane<16 row M=lane holds K=0..15; lane>=16 K=16..31
      const int kbase = hi * 16;
#pragma unroll
      for (int e = 0; e < 16; ++e) {
        int k = kbase + e;
        int t = tval[mb + k];
        a0[e] = (t == j)      ? (_Float16)1.0f : (_Float16)0.0f;
        a1[e] = (t == j + 16) ? (_Float16)1.0f : (_Float16)0.0f;
      }
      acc0 = __builtin_amdgcn_wmma_f32_16x16x32_f16(false, a0, false, bv, (short)0, acc0, false, false);
      acc1 = __builtin_amdgcn_wmma_f32_16x16x32_f16(false, a1, false, bv, (short)0, acc1, false, false);
    }
    // D layout: VGPR v holds M = v + 8*hi, N = lane&15 -> G[r, d0+j]
#pragma unroll
    for (int v = 0; v < 8; ++v) {
      int r = v + hi * 8;
      atomicAdd(&gsq[r],      acc0[v] * acc0[v]);
      atomicAdd(&gsq[r + 16], acc1[v] * acc1[v]);
    }
  }
  __syncthreads();

  // ---------------- phase 4: finalize ----------------
  if (tid < 32) {
    int r = tid;
    float c = cntf[r];
    float pairs = c * c - c;                         // == count_per_r
    float ssim  = gsq[r] - snorm[r];                 // == sum_sim_per_r
    float avg   = (pairs >= 0.5f) ? (ssim / pairs) : 0.0f;
    float term  = p_i_r[(size_t)b * RDIM + r] * (1.0f - avg);
    float sum_term = wave_sum(term);
    float amax     = wave_max(alphas[r]);
    float nv       = wave_sum(valf[r * 4] + valf[r * 4 + 1] + valf[r * 4 + 2] + valf[r * 4 + 3]);
    if (tid == 0) {
      float d_char = fmaxf(sum_term, 0.0f);
      float max_edges = nv * (nv - 1.0f) * 0.5f;
      float den = fmaxf(max_edges, 1.0f) * amax * tau_max[0];
      float d_rel = fmaxf(1.0f - s_num / fmaxf(den, 1e-12f), 0.0f);
      out[b] = w_r[0] * d_rel + w_c[0] * d_char;
    }
  }
}

extern "C" void kernel_launch(void* const* d_in, const int* in_sizes, int n_in,
                              void* d_out, int out_size, void* d_ws, size_t ws_size,
                              hipStream_t stream) {
  const float* ego_adj        = (const float*)d_in[0];
  const int*   ego_rel        = (const int*)d_in[1];
  const float* alpha_r        = (const float*)d_in[2];
  const float* tau_max        = (const float*)d_in[3];
  const float* p_i_r          = (const float*)d_in[4];
  const float* h_raw          = (const float*)d_in[5];
  const unsigned char* adj_mk = (const unsigned char*)d_in[6];
  const float* w_r            = (const float*)d_in[7];
  const float* w_c            = (const float*)d_in[8];
  float* out = (float*)d_out;

  dim3 grid(1024), block(256);
  hipLaunchKernelGGL(itds_kernel, grid, block, 0, stream,
                     ego_adj, ego_rel, alpha_r, tau_max, p_i_r,
                     h_raw, adj_mk, w_r, w_c, out);
}
